// RNN_88622355185740
// MI455X (gfx1250) — compile-verified, matrix operand order checked by reading, not run on hardware
//
#include <hip/hip_runtime.h>
#include <hip/hip_bf16.h>

// ---------------------------------------------------------------------------
// Types for CDNA5 WMMA
// ---------------------------------------------------------------------------
typedef __bf16 v16bf __attribute__((ext_vector_type(16)));
typedef float  v8f   __attribute__((ext_vector_type(8)));

union Frag { v16bf v; uint4 q[2]; };

// two f32 -> packed bf16x2, RNE, single VALU op on gfx1250
__device__ __forceinline__ unsigned pack2(float a, float b) {
    unsigned r;
    asm volatile("v_cvt_pk_bf16_f32 %0, %1, %2" : "=v"(r) : "v"(a), "v"(b));
    return r;
}
__device__ __forceinline__ float bf2f(unsigned h) {
    return __uint_as_float(h << 16);
}

// ---------------------------------------------------------------------------
// Workspace layout (bytes)
// ---------------------------------------------------------------------------
static const size_t OFF_PRE = 0;                         // 256*8192 f32 = 8 MB
static const size_t OFF_WHH = 8388608;                   // 8192*2048 bf16 = 33.5 MB
static const size_t OFF_HS  = OFF_WHH + 33554432;        // 256*2048 f32 = 2 MB
static const size_t OFF_H0  = OFF_HS + 2097152;          // 2048 f32
static const size_t OFF_C   = OFF_H0 + 8192;             // 2048 f32
static const size_t OFF_CNT = OFF_C + 8192;              // barrier counters
static const size_t OFF_H1  = OFF_CNT + 256;             // 2048 f32
static const size_t ZERO_BYTES = 8192 + 8192 + 256;      // h0, c, counters

// ---------------------------------------------------------------------------
// Pack W_hh f32 -> bf16x2 (one-time)
// ---------------------------------------------------------------------------
__global__ __launch_bounds__(256) void pack_whh_kernel(
    const float* __restrict__ src, unsigned* __restrict__ dst) {
    size_t i = (size_t)blockIdx.x * 256 + threadIdx.x;   // 8,388,608 pairs exact
    float2 v = *(const float2*)(src + 2 * i);
    dst[i] = pack2(v.x, v.y);
}

// ---------------------------------------------------------------------------
// bf16 WMMA GEMM:  C[M,N] = A[M,K] * B[N,K]^T (+ bias0 + bias1) (opt ReLU)
// fp32 inputs converted inline to bf16 (v_cvt_pk_bf16_f32), fp32 accumulate.
// Block tile 128x128, BK=32, 256 threads = 8 waves, wave tile 32x64.
// Pipelined: global->reg prefetch of chunk c+1 during compute of chunk c,
// LDS double buffered => 1 barrier per chunk, branch-free steady state.
// ---------------------------------------------------------------------------
__global__ __launch_bounds__(256) void gemm_bf16_wmma(
    const float* __restrict__ A, const float* __restrict__ B,
    const float* __restrict__ bias0, const float* __restrict__ bias1,
    float* __restrict__ C, int M, int N, int K, int relu)
{
    __shared__ unsigned short sA[2][128 * 48];  // row stride 48 bf16 = 96B
    __shared__ unsigned short sB[2][128 * 48];

    const int tid  = threadIdx.x;
    const int m0   = blockIdx.y * 128;
    const int n0   = blockIdx.x * 128;
    const int wave = tid >> 5;
    const int lane = tid & 31;
    const int wm   = (wave & 3) * 32;   // wave row base in tile
    const int wn   = (wave >> 2) * 64;  // wave col base in tile
    const int lh   = lane >> 4;         // half-wave select
    const int lm   = lane & 15;

    v8f acc[2][4];
#pragma unroll
    for (int tm = 0; tm < 2; ++tm)
#pragma unroll
        for (int tn = 0; tn < 4; ++tn)
#pragma unroll
            for (int r = 0; r < 8; ++r) acc[tm][tn][r] = 0.f;

    const int ldr = tid >> 3;        // 0..31: row within 32-row slab
    const int ldk = (tid & 7) * 4;   // 0,4,...,28

    // Per-thread global staging pointers (advance by 32 floats per chunk).
    const float* aptr = A + (size_t)(m0 + ldr) * K + ldk;
    const float* bptr = B + (size_t)(n0 + ldr) * K + ldk;
    bool bval[4];
#pragma unroll
    for (int i = 0; i < 4; ++i) bval[i] = (n0 + ldr + i * 32) < N;

    // Zero the B slots this thread owns in BOTH buffers for invalid rows,
    // then never store there again -> hot loop has no N guards at all.
#pragma unroll
    for (int i = 0; i < 4; ++i) {
        if (!bval[i]) {
            uint2 z; z.x = 0u; z.y = 0u;
            *(uint2*)((char*)sB[0] + (size_t)(ldr + i * 32) * 96 + ldk * 2) = z;
            *(uint2*)((char*)sB[1] + (size_t)(ldr + i * 32) * 96 + ldk * 2) = z;
        }
    }
    __syncthreads();

    const int nfull = K >> 5;        // number of full 32-wide K chunks
    const int ktail = K & 31;

#define LOAD_REGS(RA, RB, CHUNK)                                              \
    {                                                                         \
        _Pragma("unroll")                                                     \
        for (int i = 0; i < 4; ++i) {                                         \
            RA[i] = *(const float4*)(aptr + (size_t)i * 32 * K + (size_t)(CHUNK) * 32); \
            if (bval[i])                                                      \
                RB[i] = *(const float4*)(bptr + (size_t)i * 32 * K + (size_t)(CHUNK) * 32); \
        }                                                                     \
    }

#define STAGE(BUF, RA, RB)                                                    \
    {                                                                         \
        _Pragma("unroll")                                                     \
        for (int i = 0; i < 4; ++i) {                                         \
            uint2 pk;                                                         \
            pk.x = pack2(RA[i].x, RA[i].y);                                   \
            pk.y = pack2(RA[i].z, RA[i].w);                                   \
            *(uint2*)((char*)sA[BUF] + (size_t)(ldr + i * 32) * 96 + ldk * 2) = pk; \
            if (bval[i]) {                                                    \
                pk.x = pack2(RB[i].x, RB[i].y);                               \
                pk.y = pack2(RB[i].z, RB[i].w);                               \
                *(uint2*)((char*)sB[BUF] + (size_t)(ldr + i * 32) * 96 + ldk * 2) = pk; \
            }                                                                 \
        }                                                                     \
    }

    // Fragment byte offsets per ISA 16-bit A/B layout:
    // lane<16: K 0-7 (bytes 0..15) + K 16-23 (bytes 32..47)
    // lane>=16: K 8-15 (16..31)    + K 24-31 (48..63)
#define COMPUTE(BUF)                                                          \
    {                                                                         \
        Frag fa[2], fb[4];                                                    \
        _Pragma("unroll")                                                     \
        for (int tm = 0; tm < 2; ++tm) {                                      \
            const uint4* p = (const uint4*)((char*)sA[BUF] +                  \
                (size_t)(wm + tm * 16 + lm) * 96 + lh * 16);                  \
            fa[tm].q[0] = p[0];                                               \
            fa[tm].q[1] = p[2];                                               \
        }                                                                     \
        _Pragma("unroll")                                                     \
        for (int tn = 0; tn < 4; ++tn) {                                      \
            const uint4* p = (const uint4*)((char*)sB[BUF] +                  \
                (size_t)(wn + tn * 16 + lm) * 96 + lh * 16);                  \
            fb[tn].q[0] = p[0];                                               \
            fb[tn].q[1] = p[2];                                               \
        }                                                                     \
        _Pragma("unroll")                                                     \
        for (int tm = 0; tm < 2; ++tm)                                        \
            _Pragma("unroll")                                                 \
            for (int tn = 0; tn < 4; ++tn)                                    \
                acc[tm][tn] = __builtin_amdgcn_wmma_f32_16x16x32_bf16(        \
                    false, fa[tm].v, false, fb[tn].v,                         \
                    (short)0, acc[tm][tn], false, false);                     \
    }

    // ---- prologue: stage chunk 0 ----
    if (nfull > 0) {
        {
            float4 ra[4], rb[4];
            LOAD_REGS(ra, rb, 0)
            STAGE(0, ra, rb)
        }
        __syncthreads();

        // ---- branch-free pipelined steady state ----
        for (int c = 0; c < nfull - 1; ++c) {
            float4 ra2[4], rb2[4];
            LOAD_REGS(ra2, rb2, c + 1)
            COMPUTE(c & 1)
            STAGE((c + 1) & 1, ra2, rb2)
            __syncthreads();
        }
        // ---- peeled last full chunk ----
        COMPUTE((nfull - 1) & 1)
    }

    // ---- guarded tail chunk (K % 32) ----
    if (ktail) {
        const int kc  = nfull << 5;
        const int buf = nfull & 1;
#pragma unroll
        for (int i = 0; i < 4; ++i) {
            const int k = kc + ldk;
            float4 v = make_float4(0.f, 0.f, 0.f, 0.f);
            const float* asrc = aptr + (size_t)i * 32 * K + kc;
            if (k + 0 < K) v.x = asrc[0];
            if (k + 1 < K) v.y = asrc[1];
            if (k + 2 < K) v.z = asrc[2];
            if (k + 3 < K) v.w = asrc[3];
            uint2 pk; pk.x = pack2(v.x, v.y); pk.y = pack2(v.z, v.w);
            *(uint2*)((char*)sA[buf] + (size_t)(ldr + i * 32) * 96 + ldk * 2) = pk;

            v = make_float4(0.f, 0.f, 0.f, 0.f);
            if (bval[i]) {
                const float* bsrc = bptr + (size_t)i * 32 * K + kc;
                if (k + 0 < K) v.x = bsrc[0];
                if (k + 1 < K) v.y = bsrc[1];
                if (k + 2 < K) v.z = bsrc[2];
                if (k + 3 < K) v.w = bsrc[3];
                pk.x = pack2(v.x, v.y); pk.y = pack2(v.z, v.w);
                *(uint2*)((char*)sB[buf] + (size_t)(ldr + i * 32) * 96 + ldk * 2) = pk;
            }
        }
        __syncthreads();
        COMPUTE(buf)
    }

    // ---- epilogue: bias (+relu); C/D layout: VGPR r -> row r + 8*lh, col lm
#pragma unroll
    for (int tn = 0; tn < 4; ++tn) {
        int gcol = n0 + wn + tn * 16 + lm;
        if (gcol >= N) continue;
        float bsum = bias0 ? bias0[gcol] : 0.f;
        if (bias1) bsum += bias1[gcol];
#pragma unroll
        for (int tm = 0; tm < 2; ++tm) {
#pragma unroll
            for (int r = 0; r < 8; ++r) {
                int grow = m0 + wm + tm * 16 + r + lh * 8;
                float vv = acc[tm][tn][r] + bsum;
                if (relu && vv < 0.f) vv = 0.f;
                C[(size_t)grow * N + gcol] = vv;
            }
        }
    }
#undef LOAD_REGS
#undef STAGE
#undef COMPUTE
}

// ---------------------------------------------------------------------------
// Device-wide sense-reversal barrier (AGENT scope acquire/release so the
// compiler emits proper global_wb / global_inv cache ops across WGPs).
// ---------------------------------------------------------------------------
__device__ __forceinline__ void grid_barrier(unsigned* cnt, unsigned* gen, unsigned nblk) {
    __syncthreads();
    if (threadIdx.x == 0) {
        unsigned g = __hip_atomic_load(gen, __ATOMIC_RELAXED, __HIP_MEMORY_SCOPE_AGENT);
        unsigned prev = __hip_atomic_fetch_add(cnt, 1u, __ATOMIC_ACQ_REL, __HIP_MEMORY_SCOPE_AGENT);
        if (prev == nblk - 1) {
            __hip_atomic_store(cnt, 0u, __ATOMIC_RELAXED, __HIP_MEMORY_SCOPE_AGENT);
            __hip_atomic_store(gen, g + 1u, __ATOMIC_RELEASE, __HIP_MEMORY_SCOPE_AGENT);
        } else {
            while (__hip_atomic_load(gen, __ATOMIC_ACQUIRE, __HIP_MEMORY_SCOPE_AGENT) == g)
                __builtin_amdgcn_s_sleep(1);
        }
    }
    __syncthreads();
}

// ---------------------------------------------------------------------------
// Persistent LSTM recurrence. 64 blocks x 256 thr = 512 waves; each wave owns
// 4 h-outputs and computes all 4 gate dot-products (bf16 W_hh from L2),
// double-buffered h => one grid barrier per timestep.
// ---------------------------------------------------------------------------
__global__ __launch_bounds__(256) void lstm_seq_kernel(
    const float* __restrict__ pre, const unsigned* __restrict__ whh,
    float* __restrict__ h0, float* __restrict__ h1, float* __restrict__ cbuf,
    float* __restrict__ hs, float* __restrict__ out,
    unsigned* cnt, unsigned* gen)
{
    const int tid  = threadIdx.x;
    const int lane = tid & 31;
    const int gw   = blockIdx.x * 8 + (tid >> 5);  // 0..511
    const int n0   = gw * 4;                       // 4 outputs per wave

    for (int t = 0; t < 256; ++t) {
        const float* hcur = (t & 1) ? h1 : h0;
        float*       hnxt = (t & 1) ? h0 : h1;

        float acc[16];
#pragma unroll
        for (int d = 0; d < 16; ++d) acc[d] = 0.f;

        // K as bf16 pairs: 1024 u32 per W row; lane strides by 32.
        for (int p = lane; p < 1024; p += 32) {
            float2 hp = *(const float2*)(hcur + 2 * p);
#pragma unroll
            for (int o = 0; o < 4; ++o) {
#pragma unroll
                for (int g = 0; g < 4; ++g) {
                    unsigned w = whh[(size_t)(n0 + o + g * 2048) * 1024 + p];
                    float a = fmaf(hp.x, bf2f(w & 0xffffu), acc[o * 4 + g]);
                    acc[o * 4 + g] = fmaf(hp.y, bf2f(w >> 16), a);
                }
            }
        }
        // wave32 xor-reduction: every lane ends with all 16 totals
#pragma unroll
        for (int d = 0; d < 16; ++d) {
            float s = acc[d];
            s += __shfl_xor(s, 16, 32);
            s += __shfl_xor(s, 8, 32);
            s += __shfl_xor(s, 4, 32);
            s += __shfl_xor(s, 2, 32);
            s += __shfl_xor(s, 1, 32);
            acc[d] = s;
        }
        if (lane < 4) {
            int n = n0 + lane;
            float gi = pre[(size_t)t * 8192 + 0    + n] + acc[lane * 4 + 0];
            float gf = pre[(size_t)t * 8192 + 2048 + n] + acc[lane * 4 + 1];
            float gg = pre[(size_t)t * 8192 + 4096 + n] + acc[lane * 4 + 2];
            float go = pre[(size_t)t * 8192 + 6144 + n] + acc[lane * 4 + 3];
            float i_ = 1.f / (1.f + __expf(-gi));
            float f_ = 1.f / (1.f + __expf(-gf));
            float g_ = 1.f - 2.f / (__expf(2.f * gg) + 1.f);   // tanh
            float o_ = 1.f / (1.f + __expf(-go));
            float cv = f_ * cbuf[n] + i_ * g_;
            float tc = 1.f - 2.f / (__expf(2.f * cv) + 1.f);   // tanh
            float hv = o_ * tc;
            cbuf[n] = cv;
            hnxt[n] = hv;
            hs[(size_t)t * 2048 + n] = hv;
            if (t == 255) {
                out[(size_t)256 * 47192 + n]        = hv;  // h_n
                out[(size_t)256 * 47192 + 2048 + n] = cv;  // c_n
            }
        }
        grid_barrier(cnt, gen, 64);
    }
}

// ---------------------------------------------------------------------------
extern "C" void kernel_launch(void* const* d_in, const int* in_sizes, int n_in,
                              void* d_out, int out_size, void* d_ws, size_t ws_size,
                              hipStream_t stream) {
    const float* x     = (const float*)d_in[0];  // [256, 47192]
    const float* W_ih  = (const float*)d_in[1];  // [8192, 47192]
    const float* W_hh  = (const float*)d_in[2];  // [8192, 2048]
    const float* b_ih  = (const float*)d_in[3];  // [8192]
    const float* b_hh  = (const float*)d_in[4];  // [8192]
    const float* W_lin = (const float*)d_in[5];  // [47192, 2048]
    const float* b_lin = (const float*)d_in[6];  // [47192]
    float* out = (float*)d_out;
    char*  ws  = (char*)d_ws;

    float*    pre    = (float*)(ws + OFF_PRE);
    unsigned* whh_bf = (unsigned*)(ws + OFF_WHH);
    float*    hs     = (float*)(ws + OFF_HS);
    float*    h0     = (float*)(ws + OFF_H0);
    float*    cbuf   = (float*)(ws + OFF_C);
    unsigned* cnt    = (unsigned*)(ws + OFF_CNT);
    unsigned* gen    = cnt + 1;
    float*    h1     = (float*)(ws + OFF_H1);

    // zero h0, c0 and barrier counters (graph-capture-safe async memset)
    hipMemsetAsync(ws + OFF_H0, 0, ZERO_BYTES, stream);

    // one-time W_hh f32 -> bf16 pack (8,388,608 pairs / 256 = 32768 blocks)
    pack_whh_kernel<<<32768, 256, 0, stream>>>(W_hh, whh_bf);

    // pre = x @ W_ih^T + (b_ih + b_hh)   [256 x 8192], K = 47192
    gemm_bf16_wmma<<<dim3(64, 2), 256, 0, stream>>>(
        x, W_ih, b_ih, b_hh, pre, 256, 8192, 47192, 0);

    // sequential LSTM over 256 steps (persistent, grid-synced)
    lstm_seq_kernel<<<64, 256, 0, stream>>>(
        pre, whh_bf, h0, h1, cbuf, hs, out, cnt, gen);

    // out = relu(hs @ W_lin^T + b_lin)   [256 x 47192], K = 2048
    gemm_bf16_wmma<<<dim3(369, 2), 256, 0, stream>>>(
        hs, W_lin, b_lin, nullptr, out, 256, 47192, 2048, 1);
}
